// SpatialAttention_88347477278848
// MI455X (gfx1250) — matrix operand order repaired and checked by reading, not compiled, not run on hardware
//
#include <hip/hip_runtime.h>
#include <hip/hip_bf16.h>

typedef __attribute__((ext_vector_type(16))) _Float16 v16h;
typedef __attribute__((ext_vector_type(2)))  __fp16   v2fp16;
typedef __attribute__((ext_vector_type(8)))  float    v8f;

#define CCH   64    // channels
#define CQK   8     // q/k channels
#define BC    64    // key block (2x WMMA K)
#define WAVES 8     // waves per block; each wave owns a 16-query tile

// ---- exchange with partner lane (lane ^ 16): VALU permlanex16 if possible --
__device__ __forceinline__ unsigned xchg16_u32(unsigned v) {
#if __has_builtin(__builtin_amdgcn_permlanex16)
    return (unsigned)__builtin_amdgcn_permlanex16(
        (int)v, (int)v, 0x76543210, (int)0xFEDCBA98, false, false);
#else
    return (unsigned)__shfl_xor((int)v, 16, 32);
#endif
}
__device__ __forceinline__ float xchg16_f32(float v) {
    return __int_as_float((int)xchg16_u32((unsigned)__float_as_int(v)));
}

union PKu { v2fp16 h2; unsigned u; };
union FragU { v16h h; unsigned u[8]; };

// ---------------------------------------------------------------------------
// Projection kernel: q = wq@x+bq  (f16 [N][8]), k likewise, v = wv@x+bv
// (f16 [64][N], channel-major). One thread per spatial position.
// ---------------------------------------------------------------------------
__global__ __launch_bounds__(256) void proj_kernel(
    const float* __restrict__ x,
    const float* __restrict__ wq, const float* __restrict__ bq,
    const float* __restrict__ wk, const float* __restrict__ bk,
    const float* __restrict__ wv, const float* __restrict__ bv,
    _Float16* __restrict__ qh, _Float16* __restrict__ kh,
    _Float16* __restrict__ vh, int N)
{
    __shared__ float smem[CQK*CCH + CQK*CCH + CCH*CCH + CQK + CQK + CCH];
    float* swq = smem;
    float* swk = swq + CQK*CCH;
    float* swv = swk + CQK*CCH;
    float* sbq = swv + CCH*CCH;
    float* sbk = sbq + CQK;
    float* sbv = sbk + CQK;

    const int tid = threadIdx.x;
    for (int i = tid; i < CQK*CCH; i += 256) { swq[i] = wq[i]; swk[i] = wk[i]; }
    for (int i = tid; i < CCH*CCH; i += 256) swv[i] = wv[i];
    if (tid < CQK) { sbq[tid] = bq[tid]; sbk[tid] = bk[tid]; }
    if (tid < CCH) sbv[tid] = bv[tid];
    __syncthreads();

    const int n = blockIdx.x * 256 + tid;

    float xr[CCH];
    #pragma unroll
    for (int c = 0; c < CCH; ++c) xr[c] = x[(size_t)c * N + n];

    for (int o = 0; o < CQK; ++o) {
        float aq = sbq[o], ak = sbk[o];
        #pragma unroll
        for (int c = 0; c < CCH; ++c) {
            aq = fmaf(swq[o*CCH + c], xr[c], aq);
            ak = fmaf(swk[o*CCH + c], xr[c], ak);
        }
        qh[(size_t)n * CQK + o] = (_Float16)aq;
        kh[(size_t)n * CQK + o] = (_Float16)ak;
    }
    for (int o = 0; o < CCH; ++o) {
        float av = sbv[o];
        #pragma unroll
        for (int c = 0; c < CCH; ++c) av = fmaf(swv[o*CCH + c], xr[c], av);
        vh[(size_t)o * N + n] = (_Float16)av;
    }
}

// ---------------------------------------------------------------------------
// Flash attention, fully transposed so each lane owns ONE query column:
//   S^T = K(16k x 8c pad32) @ Q^T(32 x 16q)  -> 4x v_wmma_f32_16x16x32_f16
//   O^T = V(16c x 32k)      @ P^T(32k x 16q) -> 8x v_wmma_f32_16x16x32_f16
// Softmax is lane-local + ONE partner (lane^16) exchange; P^T B-fragments are
// assembled in registers from the lane's own packed-f16 pairs. No LDS at all.
// ---------------------------------------------------------------------------
__global__ __launch_bounds__(32 * WAVES) void attn_kernel(
    const float* __restrict__ x,
    const _Float16* __restrict__ qh,
    const _Float16* __restrict__ kh,
    const _Float16* __restrict__ vh,
    const float* __restrict__ gamma,
    float* __restrict__ out, int N)
{
    const int tid  = threadIdx.x;
    const int wave = tid >> 5;
    const int lane = tid & 31;
    const int lg   = lane >> 4;     // 0: lanes 0-15, 1: lanes 16-31
    const int ll   = lane & 15;

    const int q0 = (blockIdx.x * WAVES + wave) * 16;   // first query of tile

    // --- Q^T fragment (B layout): lane<16 = query column, elems 0..7 =
    // q-channels 0..7; channels 8..31 and lanes 16-31 are zero padding.
    v16h qb = {};
    if (lg == 0) {
        const _Float16* qp = qh + (size_t)(q0 + ll) * CQK;
        #pragma unroll
        for (int i = 0; i < 8; ++i) qb[i] = qp[i];
    }

    float mrun = -1e30f;    // running max for this lane's query
    float lpart = 0.0f;     // lane-local partial sum (rescale is row-uniform)
    v8f oacc[4] = {};       // O^T: 4 channel tiles x 8 rows, query col = ll

    for (int jb = 0; jb < N; jb += BC) {
        // --- K A-fragments: tile t = 16 keys x (8 real + 24 zero) channels -
        v16h ka[4];
        #pragma unroll
        for (int t = 0; t < 4; ++t) {
            v16h kf = {};
            if (lg == 0) {
                const _Float16* kp = kh + (size_t)(jb + t * 16 + ll) * CQK;
                #pragma unroll
                for (int i = 0; i < 8; ++i) kf[i] = kp[i];
            }
            ka[t] = kf;
        }

        // --- S^T tiles: s[t] lane holds keys (t*16 + r + 8*lg), query ll ---
        v8f s[4];
        #pragma unroll
        for (int t = 0; t < 4; ++t) {
            v8f z = {};
            s[t] = __builtin_amdgcn_wmma_f32_16x16x32_f16(
                false, ka[t], false, qb, (short)0, z, false, false);
        }

        // --- online softmax: local max tree + ONE partner exchange ---------
        float mt = s[0][0];
        #pragma unroll
        for (int t = 0; t < 4; ++t)
            #pragma unroll
            for (int r = 0; r < 8; ++r) mt = fmaxf(mt, s[t][r]);
        mt = fmaxf(mt, xchg16_f32(mt));          // full row (query) max
        const float mn = fmaxf(mrun, mt);
        const float sc = __expf(mrun - mn);      // row-uniform rescale
        mrun = mn;

        unsigned pk[4][4];                       // P as packed f16 pairs
        float psum = 0.0f;
        #pragma unroll
        for (int t = 0; t < 4; ++t) {
            #pragma unroll
            for (int j = 0; j < 4; ++j) {
                const float e0 = __expf(s[t][2*j]     - mn);
                const float e1 = __expf(s[t][2*j + 1] - mn);
                psum += e0 + e1;
                PKu p; p.h2 = __builtin_amdgcn_cvt_pkrtz(e0, e1);
                pk[t][j] = p.u;
            }
        }
        lpart = lpart * sc + psum;
        #pragma unroll
        for (int ct = 0; ct < 4; ++ct)
            #pragma unroll
            for (int r = 0; r < 8; ++r) oacc[ct][r] *= sc;

        // --- O^T += V @ P^T : 4 channel tiles x 2 key chunks ---------------
        #pragma unroll
        for (int kc = 0; kc < 2; ++kc) {
            // assemble P^T B-fragment from this lane's own P tiles 2*kc, 2*kc+1
            FragU pf;
            #pragma unroll
            for (int j = 0; j < 4; ++j) {
                const unsigned a = pk[kc*2][j];       // keys kc*32 + 0..15
                const unsigned b = pk[kc*2 + 1][j];   // keys kc*32 + 16..31
                pf.u[j]     = a;        // elems 0..7
                pf.u[4 + j] = b;        // elems 8..15
            }
            #pragma unroll
            for (int ct = 0; ct < 4; ++ct) {
                FragU vf;
                const _Float16* vp = vh + (size_t)(ct * 16 + ll) * N
                                        + jb + kc * 32 + lg * 8;
                #pragma unroll
                for (int i = 0; i < 8; ++i) {
                    vf.h[i]     = vp[i];        // K = kc*32 + lg*8 + i
                    vf.h[8 + i] = vp[16 + i];   // K = kc*32 + 16 + lg*8 + i
                }
                oacc[ct] = __builtin_amdgcn_wmma_f32_16x16x32_f16(
                    false, vf.h, false, pf.h, (short)0, oacc[ct], false, false);
            }
        }

        // hint the next key block into cache (global_prefetch_b8)
        if (jb + BC < N) {
            __builtin_prefetch(kh + (size_t)(jb + BC) * CQK, 0, 0);
            __builtin_prefetch(vh + (size_t)(ll) * N + jb + BC, 0, 0);
        }
    }

    // --- epilogue: one partner exchange finishes the row sum ---------------
    const float lrow = lpart + xchg16_f32(lpart);
    const float g = gamma[0];
    const int n = q0 + ll;                      // this lane's query position
    #pragma unroll
    for (int ct = 0; ct < 4; ++ct) {
        #pragma unroll
        for (int r = 0; r < 8; ++r) {
            const int ch = ct * 16 + r + 8 * lg;
            const float val = oacc[ct][r] / lrow;
            out[(size_t)ch * N + n] = x[(size_t)ch * N + n] + g * val;
        }
    }
}

// ---------------------------------------------------------------------------
extern "C" void kernel_launch(void* const* d_in, const int* in_sizes, int n_in,
                              void* d_out, int out_size, void* d_ws, size_t ws_size,
                              hipStream_t stream) {
    const float* x     = (const float*)d_in[0];
    const float* wq    = (const float*)d_in[1];
    const float* bq    = (const float*)d_in[2];
    const float* wk    = (const float*)d_in[3];
    const float* bk    = (const float*)d_in[4];
    const float* wv    = (const float*)d_in[5];
    const float* bv    = (const float*)d_in[6];
    const float* gamma = (const float*)d_in[7];
    float* out = (float*)d_out;

    const int N = in_sizes[0] / CCH;   // 16384 spatial positions

    char* ws = (char*)d_ws;
    _Float16* qh = (_Float16*)ws;                                   // [N][8]
    _Float16* kh = (_Float16*)(ws + (size_t)N * CQK * 2);           // [N][8]
    _Float16* vh = (_Float16*)(ws + (size_t)N * CQK * 4);           // [64][N]

    proj_kernel<<<N / 256, 256, 0, stream>>>(x, wq, bq, wk, bk, wv, bv,
                                             qh, kh, vh, N);
    attn_kernel<<<N / (16 * WAVES), 32 * WAVES, 0, stream>>>(
        x, qh, kh, vh, gamma, out, N);
}
